// GGNNEncoder_25194278158853
// MI455X (gfx1250) — matrix operand order repaired and checked by reading, not compiled, not run on hardware
//
#include <hip/hip_runtime.h>
#include <hip/hip_bf16.h>

// ---------------------------------------------------------------------------
// GGNN encoder for MI455X (gfx1250): bf16 WMMA GEMMs + L2-resident f32 scatter
// ---------------------------------------------------------------------------

#define N_NODES      100000
#define HIDDEN       128
#define N_EDGES      400000
#define N_ETYPES     6          // 2 * N_EDGE_TYPES (fwd + reversed)
#define GNN_LAYERS   8
#define SMALL_NUM    1e-8f

typedef __attribute__((ext_vector_type(16))) __bf16 v16bf;
typedef __attribute__((ext_vector_type(8)))  float  v8f;

// ---- fragment loaders (CDNA5 wave32 WMMA; A and B use contiguous k = 16*(lane/16) + elem) --------

// A tile 16x32 bf16, row-major source rows of length `stride`.
// lane 0-15 : M=lane, elems 0..15 -> K=k0+0..15
// lane16-31 : M=lane-16, elems 0..15 -> K=k0+16..31
__device__ inline v16bf load_a_frag(const __bf16* __restrict__ base, int stride,
                                    int lane, int k0) {
    int m  = lane & 15;
    int kh = (lane >> 4) & 1;
    const __bf16* p0 = base + m * stride + k0 + kh * 16;
    const __bf16* p1 = p0 + 8;
    v16bf a;
#pragma unroll
    for (int i = 0; i < 8; ++i) { a[i] = p0[i]; a[8 + i] = p1[i]; }
    return a;
}

// B tile 32x16 bf16 given as transposed storage BT[n][k] (contiguous k).
// b[i] = B[k0 + g*16 + i][n] = BT[n*stride + k0 + g*16 + i]
__device__ inline v16bf load_b_fragT(const __bf16* __restrict__ baseT, int stride,
                                     int lane, int k0) {
    int n = lane & 15;
    int g = (lane >> 4) & 1;
    const __bf16* p = baseT + n * stride + k0 + g * 16;
    v16bf b;
#pragma unroll
    for (int i = 0; i < 16; ++i) b[i] = p[i];
    return b;
}

__device__ inline float fast_sigmoid(float x) {
    return 1.0f / (1.0f + __expf(-x));
}

// ---- one-time weight conversion -------------------------------------------
// W_msg[e][d][h] -> wmsgT bf16 [e][h][d]  (so B-fragments read contiguous K=d)
// w_ih / w_hh [384][128] -> bf16 as-is    (rows already contiguous in K)
__global__ void cvt_weights_kernel(const float* __restrict__ W_msg,
                                   const float* __restrict__ w_ih,
                                   const float* __restrict__ w_hh,
                                   __bf16* __restrict__ wmsgT,
                                   __bf16* __restrict__ wihbf,
                                   __bf16* __restrict__ whhbf) {
    int idx = blockIdx.x * blockDim.x + threadIdx.x;
    if (idx < N_ETYPES * HIDDEN * HIDDEN) {
        int e = idx >> 14;
        int d = (idx >> 7) & 127;
        int h = idx & 127;
        wmsgT[(e << 14) + (h << 7) + d] = (__bf16)W_msg[idx];
    }
    if (idx < 3 * HIDDEN * HIDDEN) {
        wihbf[idx] = (__bf16)w_ih[idx];
        whhbf[idx] = (__bf16)w_hh[idx];
    }
}

// ---- layer-invariant in-degree counts (edge list is fixed) ----------------
__global__ void count_kernel(const int* __restrict__ edges,
                             float* __restrict__ counts) {
    int idx = blockIdx.x * blockDim.x + threadIdx.x;
    if (idx >= 3 * N_EDGES) return;
    int a = edges[idx * 2 + 0];
    int b = edges[idx * 2 + 1];
    atomicAdd(&counts[b], 1.0f);   // forward edge  : tgt = b
    atomicAdd(&counts[a], 1.0f);   // reversed edge : tgt = a
}

// counts -> reciprocal divisor, in place: rdiv = 1/max(c,1)
__global__ void rdiv_kernel(float* __restrict__ counts) {
    int n = blockIdx.x * blockDim.x + threadIdx.x;
    if (n >= N_NODES) return;
    float c = counts[n];
    counts[n] = (c == 0.0f) ? 1.0f : (1.0f / c);
}

// ---- f32 -> bf16 (vectorized x4; layer-0 seed only) -----------------------
__global__ void cvt_bf16_kernel(const float* __restrict__ in,
                                __bf16* __restrict__ out, int total4) {
    int idx = blockIdx.x * blockDim.x + threadIdx.x;
    if (idx >= total4) return;
    const float4 v = ((const float4*)in)[idx];
    __bf16* o = out + idx * 4;
    o[0] = (__bf16)v.x; o[1] = (__bf16)v.y; o[2] = (__bf16)v.z; o[3] = (__bf16)v.w;
}

// ---- message GEMM: prop[e] = h @ W_msg[e] + b_msg[e], all 6 e per wave ----
// one wave -> 6 x (16x16) output tiles sharing the A fragment; 24 WMMAs
__global__ void gemm_msg_kernel(const __bf16* __restrict__ hbf,
                                const __bf16* __restrict__ wmsgT,
                                const float* __restrict__ b_msg,
                                float* __restrict__ prop) {
    int wave = (blockIdx.x * blockDim.x + threadIdx.x) >> 5;
    int lane = threadIdx.x & 31;
    int nc   = wave & 7;                 // 8 column tiles of 16
    int mt   = wave >> 3;                // node tile
    if (mt >= N_NODES / 16) return;      // wave-uniform
    int row0 = mt * 16, col0 = nc * 16;
    int n  = lane & 15;
    int mh = (lane >> 4) * 8;

    v8f acc[N_ETYPES];
#pragma unroll
    for (int e = 0; e < N_ETYPES; ++e) {
        float bias = b_msg[e * HIDDEN + col0 + n];
#pragma unroll
        for (int i = 0; i < 8; ++i) acc[e][i] = bias;
    }

    const __bf16* aB = hbf + (size_t)row0 * HIDDEN;
#pragma unroll
    for (int k0 = 0; k0 < HIDDEN; k0 += 32) {
        v16bf a = load_a_frag(aB, HIDDEN, lane, k0);
#pragma unroll
        for (int e = 0; e < N_ETYPES; ++e) {
            v16bf b = load_b_fragT(wmsgT + ((size_t)e * HIDDEN + col0) * HIDDEN,
                                   HIDDEN, lane, k0);
            acc[e] = __builtin_amdgcn_wmma_f32_16x16x32_bf16(false, a, false, b,
                                                             (short)0, acc[e],
                                                             false, false);
        }
    }

#pragma unroll
    for (int e = 0; e < N_ETYPES; ++e) {
        float* out = prop + (size_t)e * N_NODES * HIDDEN;
#pragma unroll
        for (int i = 0; i < 8; ++i)
            out[(size_t)(row0 + mh + i) * HIDDEN + col0 + n] = acc[e][i];
    }
}

// ---- edge scatter: msgs[tgt] += prop[e][src] ------------------------------
// one wave per edge; msgs (51 MB) is L2-resident so atomics stay on-chip
__global__ void scatter_kernel(const float* __restrict__ prop,
                               const int* __restrict__ edges,
                               float* __restrict__ msgs) {
    int wave = (blockIdx.x * blockDim.x + threadIdx.x) >> 5;
    int lane = threadIdx.x & 31;
    if (wave >= N_ETYPES * N_EDGES) return;   // wave-uniform
    int e = wave / N_EDGES;
    int m = wave - e * N_EDGES;
    int et = (e < 3) ? e : e - 3;
    const int* ep = edges + ((size_t)et * N_EDGES + m) * 2;
    int a = ep[0], b = ep[1];
    int src = (e < 3) ? a : b;
    int tgt = (e < 3) ? b : a;

    const float4 v = ((const float4*)(prop + ((size_t)e * N_NODES + src) * HIDDEN))[lane];
    float* drow = msgs + (size_t)tgt * HIDDEN + lane * 4;
    atomicAdd(drow + 0, v.x);
    atomicAdd(drow + 1, v.y);
    atomicAdd(drow + 2, v.z);
    atomicAdd(drow + 3, v.w);
}

// ---- normalize messages, add epsilon, emit bf16 x (x4 vectorized) ---------
__global__ void prep_x_kernel(const float* __restrict__ msgs,
                              const float* __restrict__ rdiv,
                              __bf16* __restrict__ xbf) {
    int idx = blockIdx.x * blockDim.x + threadIdx.x;   // quad index
    if (idx >= N_NODES * HIDDEN / 4) return;
    float r = rdiv[idx >> 5];                          // node = idx*4 / 128
    const float4 v = ((const float4*)msgs)[idx];
    __bf16* o = xbf + idx * 4;
    o[0] = (__bf16)__builtin_fmaf(v.x, r, SMALL_NUM);
    o[1] = (__bf16)__builtin_fmaf(v.y, r, SMALL_NUM);
    o[2] = (__bf16)__builtin_fmaf(v.z, r, SMALL_NUM);
    o[3] = (__bf16)__builtin_fmaf(v.w, r, SMALL_NUM);
}

// ---- fused GRU: gi = x@w_ih^T, gh = h@w_hh^T, gates, blend ----------------
// one wave -> one 16x16 tile of h_new; 24 WMMAs; emits f32 + bf16 h_next
__global__ void gru_kernel(const __bf16* __restrict__ xbf,
                           const __bf16* __restrict__ hbf,
                           const float* __restrict__ hin,
                           const __bf16* __restrict__ wihbf,
                           const __bf16* __restrict__ whhbf,
                           const float* __restrict__ b_ih,
                           const float* __restrict__ b_hh,
                           float* __restrict__ hout,
                           __bf16* __restrict__ houtbf) {
    int lane = threadIdx.x & 31;
    int nc   = threadIdx.x >> 5;         // 0..7
    int mt   = blockIdx.x;               // 0..6249
    int row0 = mt * 16, col0 = nc * 16;
    int n = lane & 15;

    v8f ir, iz, in_, hr, hz, hn;
    {
        float b0 = b_ih[            col0 + n];
        float b1 = b_ih[  HIDDEN  + col0 + n];
        float b2 = b_ih[2*HIDDEN  + col0 + n];
        float b3 = b_hh[            col0 + n];
        float b4 = b_hh[  HIDDEN  + col0 + n];
        float b5 = b_hh[2*HIDDEN  + col0 + n];
#pragma unroll
        for (int i = 0; i < 8; ++i) {
            ir[i] = b0; iz[i] = b1; in_[i] = b2;
            hr[i] = b3; hz[i] = b4; hn[i]  = b5;
        }
    }

    const __bf16* aX = xbf + (size_t)row0 * HIDDEN;
    const __bf16* aH = hbf + (size_t)row0 * HIDDEN;
    const __bf16* bir = wihbf + (size_t)(0 * HIDDEN + col0) * HIDDEN;
    const __bf16* biz = wihbf + (size_t)(1 * HIDDEN + col0) * HIDDEN;
    const __bf16* bin = wihbf + (size_t)(2 * HIDDEN + col0) * HIDDEN;
    const __bf16* bhr = whhbf + (size_t)(0 * HIDDEN + col0) * HIDDEN;
    const __bf16* bhz = whhbf + (size_t)(1 * HIDDEN + col0) * HIDDEN;
    const __bf16* bhn = whhbf + (size_t)(2 * HIDDEN + col0) * HIDDEN;

#pragma unroll
    for (int k0 = 0; k0 < HIDDEN; k0 += 32) {
        v16bf ax = load_a_frag(aX, HIDDEN, lane, k0);
        v16bf ah = load_a_frag(aH, HIDDEN, lane, k0);
        v16bf b;
        b = load_b_fragT(bir, HIDDEN, lane, k0);
        ir  = __builtin_amdgcn_wmma_f32_16x16x32_bf16(false, ax, false, b, (short)0, ir,  false, false);
        b = load_b_fragT(biz, HIDDEN, lane, k0);
        iz  = __builtin_amdgcn_wmma_f32_16x16x32_bf16(false, ax, false, b, (short)0, iz,  false, false);
        b = load_b_fragT(bin, HIDDEN, lane, k0);
        in_ = __builtin_amdgcn_wmma_f32_16x16x32_bf16(false, ax, false, b, (short)0, in_, false, false);
        b = load_b_fragT(bhr, HIDDEN, lane, k0);
        hr  = __builtin_amdgcn_wmma_f32_16x16x32_bf16(false, ah, false, b, (short)0, hr,  false, false);
        b = load_b_fragT(bhz, HIDDEN, lane, k0);
        hz  = __builtin_amdgcn_wmma_f32_16x16x32_bf16(false, ah, false, b, (short)0, hz,  false, false);
        b = load_b_fragT(bhn, HIDDEN, lane, k0);
        hn  = __builtin_amdgcn_wmma_f32_16x16x32_bf16(false, ah, false, b, (short)0, hn,  false, false);
    }

    int mh = (lane >> 4) * 8;
#pragma unroll
    for (int i = 0; i < 8; ++i) {
        size_t off = (size_t)(row0 + mh + i) * HIDDEN + col0 + n;
        float hv = hin[off];
        float r  = fast_sigmoid(ir[i] + hr[i]);
        float z  = fast_sigmoid(iz[i] + hz[i]);
        float nn = tanhf(in_[i] + r * hn[i]);
        float hv2 = (1.0f - z) * nn + z * hv;
        hout[off]   = hv2;
        houtbf[off] = (__bf16)hv2;
    }
}

// ---------------------------------------------------------------------------
extern "C" void kernel_launch(void* const* d_in, const int* in_sizes, int n_in,
                              void* d_out, int out_size, void* d_ws, size_t ws_size,
                              hipStream_t stream) {
    (void)in_sizes; (void)n_in; (void)out_size; (void)ws_size;

    const float* node_states = (const float*)d_in[0];
    const int*   edges       = (const int*)d_in[1];
    const float* W_msg       = (const float*)d_in[2];
    const float* b_msg       = (const float*)d_in[3];
    const float* w_ih        = (const float*)d_in[4];
    const float* w_hh        = (const float*)d_in[5];
    const float* b_ih        = (const float*)d_in[6];
    const float* b_hh        = (const float*)d_in[7];

    // ---- workspace layout (~490 MB) ----
    char* ws = (char*)d_ws;
    const size_t PROP_B  = (size_t)N_ETYPES * N_NODES * HIDDEN * 4;  // 307.2 MB
    const size_t MSGS_B  = (size_t)N_NODES * HIDDEN * 4;             //  51.2 MB
    const size_t CNT_B   = (size_t)N_NODES * 4;
    const size_t HBF_B   = (size_t)N_NODES * HIDDEN * 2;

    size_t off = 0;
    float*  prop   = (float*)(ws + off);  off += PROP_B;
    float*  msgs   = (float*)(ws + off);  off += MSGS_B;
    float*  rdiv   = (float*)(ws + off);  off += CNT_B;
    off = (off + 255) & ~(size_t)255;
    float*  hA     = (float*)(ws + off);  off += MSGS_B;
    __bf16* hbfA   = (__bf16*)(ws + off); off += HBF_B;
    __bf16* hbfB   = (__bf16*)(ws + off); off += HBF_B;
    __bf16* xbf    = (__bf16*)(ws + off); off += HBF_B;
    __bf16* wmsgT  = (__bf16*)(ws + off); off += (size_t)N_ETYPES * HIDDEN * HIDDEN * 2;
    __bf16* wihbf  = (__bf16*)(ws + off); off += (size_t)3 * HIDDEN * HIDDEN * 2;
    __bf16* whhbf  = (__bf16*)(ws + off); off += (size_t)3 * HIDDEN * HIDDEN * 2;

    float* outH = (float*)d_out;  // first half: final h

    // second half of d_out: copy of original node_states
    hipMemcpyAsync(outH + (size_t)N_NODES * HIDDEN, node_states,
                   (size_t)N_NODES * HIDDEN * 4, hipMemcpyDeviceToDevice, stream);

    // one-time prep: weights -> bf16, in-degree reciprocals, layer-0 h -> bf16
    cvt_weights_kernel<<<(N_ETYPES * HIDDEN * HIDDEN + 255) / 256, 256, 0, stream>>>(
        W_msg, w_ih, w_hh, wmsgT, wihbf, whhbf);
    hipMemsetAsync(rdiv, 0, CNT_B, stream);
    count_kernel<<<(3 * N_EDGES + 255) / 256, 256, 0, stream>>>(edges, rdiv);
    rdiv_kernel<<<(N_NODES + 255) / 256, 256, 0, stream>>>(rdiv);
    cvt_bf16_kernel<<<(N_NODES * HIDDEN / 4 + 255) / 256, 256, 0, stream>>>(
        node_states, hbfA, N_NODES * HIDDEN / 4);

    const float* hin = node_states;
    __bf16* hbf_cur = hbfA;
    for (int layer = 0; layer < GNN_LAYERS; ++layer) {
        float*  hout     = (layer & 1) ? outH : hA;   // layer 7 (last) -> d_out
        __bf16* hbf_next = (layer & 1) ? hbfA : hbfB;

        // prop[e] = h @ W_msg[e] + b_msg[e]   (all 6 e per wave)
        gemm_msg_kernel<<<N_NODES / 16 * 8 / 8, 256, 0, stream>>>(
            hbf_cur, wmsgT, b_msg, prop);

        // zero msgs, then scatter edge messages
        hipMemsetAsync(msgs, 0, MSGS_B, stream);
        scatter_kernel<<<N_ETYPES * N_EDGES / 8, 256, 0, stream>>>(prop, edges, msgs);

        // x = msgs * rdiv + eps  -> bf16
        prep_x_kernel<<<(N_NODES * HIDDEN / 4 + 255) / 256, 256, 0, stream>>>(
            msgs, rdiv, xbf);

        // fused GRU update (emits f32 h and bf16 h for the next layer)
        gru_kernel<<<N_NODES / 16, 256, 0, stream>>>(
            xbf, hbf_cur, hin, wihbf, whhbf, b_ih, b_hh, hout, hbf_next);

        hin = hout;
        hbf_cur = hbf_next;
    }
}
